// PointNet2Backbone_83425444757835
// MI455X (gfx1250) — matrix operand order repaired and checked by reading, not compiled, not run on hardware
//
#include <hip/hip_runtime.h>
#include <cstdint>
#include <cstddef>

typedef __attribute__((ext_vector_type(16))) __bf16 v16bf;
typedef __attribute__((ext_vector_type(8)))  __bf16 v8bf;
typedef __attribute__((ext_vector_type(8)))  float  v8f;

#define BN_EPS 1e-5f
#define TILE_K 32

// ---------------- Farthest point sampling (1 block per batch) ----------------
__global__ __launch_bounds__(1024)
void fps_kernel(const float* __restrict__ xyz, float* __restrict__ nxyz,
                int N, int npoint)
{
  __shared__ float dist[4096];
  __shared__ float rv[1024];
  __shared__ int   ri[1024];
  const int b = blockIdx.x;
  const int tid = threadIdx.x;
  const int bs = blockDim.x;
  const float* base = xyz + (size_t)b * N * 3;
  for (int i = tid; i < N; i += bs) dist[i] = 1e10f;
  __syncthreads();
  int far = 0;
  for (int it = 0; it < npoint; ++it) {
    float cx = base[far*3+0], cy = base[far*3+1], cz = base[far*3+2];
    if (tid == 0) {
      float* o = nxyz + ((size_t)b * npoint + it) * 3;
      o[0] = cx; o[1] = cy; o[2] = cz;
    }
    float best = -1.f; int bi = 0;
    for (int i = tid; i < N; i += bs) {
      float dx = base[i*3+0]-cx, dy = base[i*3+1]-cy, dz = base[i*3+2]-cz;
      float d = dx*dx + dy*dy + dz*dz;
      float nd = fminf(dist[i], d);
      dist[i] = nd;
      if (nd > best) { best = nd; bi = i; }   // keeps earliest index on ties
    }
    rv[tid] = best; ri[tid] = bi;
    __syncthreads();
    for (int st = bs >> 1; st > 0; st >>= 1) {
      if (tid < st) {
        float ov = rv[tid+st]; int oi = ri[tid+st];
        if (ov > rv[tid] || (ov == rv[tid] && oi < ri[tid])) { rv[tid]=ov; ri[tid]=oi; }
      }
      __syncthreads();
    }
    far = ri[0];
    __syncthreads();
  }
}

// ---------------- Ball query: one wave32 per centroid ----------------
__global__ __launch_bounds__(256)
void ball_query_kernel(const float* __restrict__ xyz, const float* __restrict__ nxyz,
                       int* __restrict__ out, int B, int N, int S, int K, float r2)
{
  const int gw = (int)((blockIdx.x * blockDim.x + threadIdx.x) >> 5);
  const int lane = threadIdx.x & 31;
  if (gw >= B * S) return;
  const int b = gw / S;
  const float* base = xyz + (size_t)b * N * 3;
  const float cx = nxyz[(size_t)gw*3+0];
  const float cy = nxyz[(size_t)gw*3+1];
  const float cz = nxyz[(size_t)gw*3+2];
  int* o = out + (size_t)gw * K;
  int cnt = 0;
  int first = -1;
  for (int bidx = 0; bidx < N && cnt < K; bidx += 32) {
    const int i = bidx + lane;
    bool ok = false;
    if (i < N) {
      float dx = base[i*3+0]-cx, dy = base[i*3+1]-cy, dz = base[i*3+2]-cz;
      ok = (dx*dx + dy*dy + dz*dz) <= r2;
    }
    unsigned mask = __builtin_amdgcn_ballot_w32(ok);
    if (first < 0 && mask) first = bidx + __builtin_ctz(mask);
    int pos = cnt + __popc(mask & ((1u << lane) - 1u));
    if (ok && pos < K) o[pos] = i;
    cnt += __popc(mask);
  }
  if (first < 0) first = 0;
  if (cnt > K) cnt = K;
  for (int p = cnt + lane; p < K; p += 32) o[p] = first;   // fill with first idx
}

// ---------------- Gather + center + bf16 pack (zero-padded channels) --------
__global__ __launch_bounds__(64)
void build_group_kernel(const float* __restrict__ xyz, const float* __restrict__ pts,
                        const float* __restrict__ nxyz, const int* __restrict__ idx,
                        __bf16* __restrict__ X, int N, int S, int K, int Cprev, int Cpad)
{
  const int r = blockIdx.x;           // row m = (b*S+s)*K + k
  const int k = r % K;
  const int bs_ = r / K;
  const int b = bs_ / S;
  const int id = idx[(size_t)bs_ * K + k];
  const float* src = xyz + ((size_t)b * N + id) * 3;
  const float* ctr = nxyz + (size_t)bs_ * 3;
  __bf16* xr = X + (size_t)r * Cpad;
  for (int c = threadIdx.x; c < Cpad; c += 64) {
    float v = 0.f;
    if (c < Cprev)            v = pts[((size_t)b * N + id) * Cprev + c];
    else if (c < Cprev + 3)   { int d = c - Cprev; v = src[d] - ctr[d]; }
    xr[c] = (__bf16)v;
  }
}

// ---------------- Weight f32 -> bf16 with K padding ----------------
__global__ __launch_bounds__(128)
void convert_w_kernel(const float* __restrict__ W, __bf16* __restrict__ Wb,
                      int cin, int Kp)
{
  const int o = blockIdx.x;
  for (int c = threadIdx.x; c < Kp; c += 128)
    Wb[(size_t)o * Kp + c] = (c < cin) ? (__bf16)W[(size_t)o * cin + c] : (__bf16)0.f;
}

// ---------------- WMMA bf16 GEMM: Y[M,Cout] = X[M,Kp] * Wb[Cout,Kp]^T + bias -
// Block: 256 threads = 8 waves. Block tile 128(M) x 32(N); each wave owns a
// 16(M) x 32(N) macro-tile = 2 WMMA accumulators sharing one A fragment, so
// A is fetched from global exactly once (no cross-wave redundancy).
__global__ __launch_bounds__(256)
void gemm_bf16_wmma(const __bf16* __restrict__ X, const __bf16* __restrict__ Wb,
                    const float* __restrict__ bias, float* __restrict__ Y,
                    int M, int Kp, int Cout)
{
  __shared__ __bf16 Bt[TILE_K][40];        // [k][n], padded stride 40 halves
  const int tid  = threadIdx.x;
  const int wave = tid >> 5;               // 0..7 -> 8 M-subtiles
  const int lane = tid & 31;
  const int m0     = blockIdx.x * 128 + wave * 16;
  const int nBlock = blockIdx.y * 32;

  v8f acc0 = {0.f,0.f,0.f,0.f,0.f,0.f,0.f,0.f};
  v8f acc1 = {0.f,0.f,0.f,0.f,0.f,0.f,0.f,0.f};

  // A-fragment layout (ISA 7.12.2, 16-bit A 16x32):
  //   lanes 0-15 : row M=lane,    elems 0-7 = K 0..7,  elems 8-15 = K 16..23
  //   lanes 16-31: row M=lane-16, elems 0-7 = K 8..15, elems 8-15 = K 24..31
  const int rowA = m0 + (lane & 15);
  const int koff = (lane >> 4) * 8;
  const __bf16* xrow = X + (size_t)rowA * Kp;

  const int wn = tid >> 3;                 // staging: n row of W tile (0..31)
  const int wk = (tid & 7) * 4;            // 4 halves along K per thread

  for (int k0 = 0; k0 < Kp; k0 += TILE_K) {
    // Stage B tile transposed into LDS: Bt[k][n] = Wb[nBlock+n][k0+k]
    {
      const __bf16* wr = Wb + (size_t)(nBlock + wn) * Kp + k0 + wk;
      __bf16 h0 = wr[0], h1 = wr[1], h2 = wr[2], h3 = wr[3];
      Bt[wk+0][wn] = h0; Bt[wk+1][wn] = h1; Bt[wk+2][wn] = h2; Bt[wk+3][wn] = h3;
    }
    __syncthreads();

    if (k0 + 2*TILE_K < Kp) __builtin_prefetch(xrow + k0 + 2*TILE_K, 0, 1);

    v8bf alo = *(const v8bf*)(xrow + k0 + koff);
    v8bf ahi = *(const v8bf*)(xrow + k0 + 16 + koff);
    v16bf a;
#pragma unroll
    for (int j = 0; j < 8; ++j) { a[j] = alo[j]; a[8+j] = ahi[j]; }

    // B-fragments: lane&15 = N column; elems follow A's K order (koff+j, 16+koff+j)
    const int bn = lane & 15;
    v16bf bf0, bf1;
#pragma unroll
    for (int j = 0; j < 8; ++j) {
      bf0[j] = Bt[koff+j][bn];    bf0[8+j] = Bt[16+koff+j][bn];
      bf1[j] = Bt[koff+j][bn+16]; bf1[8+j] = Bt[16+koff+j][bn+16];
    }

    acc0 = __builtin_amdgcn_wmma_f32_16x16x32_bf16(
        false, a, false, bf0, (short)0, acc0, false, false);
    acc1 = __builtin_amdgcn_wmma_f32_16x16x32_bf16(
        false, a, false, bf1, (short)0, acc1, false, false);
    __syncthreads();
  }

  // D layout: elem j -> row m0 + (lane>=16 ? 8 : 0) + j, col = n0 + (lane&15)
  const int col0 = nBlock + (lane & 15);
  const int col1 = col0 + 16;
  const float bb0 = bias[col0];
  const float bb1 = bias[col1];
  const int mrow = m0 + (lane >> 4) * 8;
#pragma unroll
  for (int j = 0; j < 8; ++j) {
    Y[(size_t)(mrow + j) * Cout + col0] = acc0[j] + bb0;
    Y[(size_t)(mrow + j) * Cout + col1] = acc1[j] + bb1;
  }
}

// ---------------- per-channel batch stats over M rows ----------------
__global__ __launch_bounds__(256)
void stats_kernel(const float* __restrict__ Y, int M, int C,
                  float* __restrict__ meanv, float* __restrict__ rstdv)
{
  __shared__ float s1[256];
  __shared__ float s2[256];
  const int c = blockIdx.x;
  const int tid = threadIdx.x;
  float a = 0.f, q = 0.f;
  for (int m = tid; m < M; m += 256) {
    float v = Y[(size_t)m * C + c];
    a += v; q += v * v;
  }
  s1[tid] = a; s2[tid] = q;
  __syncthreads();
  for (int st = 128; st > 0; st >>= 1) {
    if (tid < st) { s1[tid] += s1[tid+st]; s2[tid] += s2[tid+st]; }
    __syncthreads();
  }
  if (tid == 0) {
    float inv = 1.f / (float)M;
    float mean = s1[0] * inv;
    float var  = s2[0] * inv - mean * mean;
    if (var < 0.f) var = 0.f;
    meanv[c] = mean;
    rstdv[c] = rsqrtf(var + BN_EPS);
  }
}

// ---------------- BN + ReLU -> bf16 (next layer input) ----------------
__global__ __launch_bounds__(256)
void bn_relu_bf16_kernel(const float* __restrict__ Y, const float* __restrict__ meanv,
                         const float* __restrict__ rstdv, const float* __restrict__ gamma,
                         const float* __restrict__ beta, __bf16* __restrict__ X,
                         size_t total, int C)
{
  size_t i = (size_t)blockIdx.x * 256 + threadIdx.x;
  if (i >= total) return;
  int c = (int)(i % C);
  float v = (Y[i] - meanv[c]) * rstdv[c] * gamma[c] + beta[c];
  X[i] = (__bf16)(v > 0.f ? v : 0.f);
}

// ---------------- BN + ReLU + max over K neighbors -> concat output ---------
__global__ __launch_bounds__(256)
void bn_relu_maxpool_kernel(const float* __restrict__ Y, const float* __restrict__ meanv,
                            const float* __restrict__ rstdv, const float* __restrict__ gamma,
                            const float* __restrict__ beta, float* __restrict__ outp,
                            int BS, int K, int C, int coff, int Ctot)
{
  size_t i = (size_t)blockIdx.x * 256 + threadIdx.x;
  if (i >= (size_t)BS * C) return;
  int bs = (int)(i / C);
  int c  = (int)(i % C);
  float mn = meanv[c], rs = rstdv[c], gm = gamma[c], bt = beta[c];
  float mx = -1e30f;
  const float* yr = Y + (size_t)bs * K * C + c;
  for (int k = 0; k < K; ++k) {
    float v = (yr[(size_t)k * C] - mn) * rs * gm + bt;
    v = v > 0.f ? v : 0.f;
    mx = fmaxf(mx, v);
  }
  outp[(size_t)bs * Ctot + coff + c] = mx;
}

// ---------------- Projection head ----------------
__global__ __launch_bounds__(256)
void proj_gemm_kernel(const float* __restrict__ feats, const float* __restrict__ W,
                      const float* __restrict__ bias, float* __restrict__ y,
                      int Cin, int Cout, int total)
{
  int i = blockIdx.x * 256 + threadIdx.x;
  if (i >= total) return;
  int b  = i / Cout;
  int co = i % Cout;
  const float* f = feats + (size_t)b * Cin;
  const float* w = W + (size_t)co * Cin;
  float s = bias[co];
  for (int c = 0; c < Cin; ++c) s += f[c] * w[c];
  y[i] = s;
}

__global__ __launch_bounds__(16)
void proj_bn_kernel(const float* __restrict__ y, const float* __restrict__ gamma,
                    const float* __restrict__ beta, float* __restrict__ out)
{
  __shared__ float sv[16];
  const int co = blockIdx.x;
  const int b = threadIdx.x;            // 16 threads = 16 batch rows
  float v = y[(size_t)b * 512 + co];
  sv[b] = v;
  __syncthreads();
  float s = 0.f, q = 0.f;
  for (int i = 0; i < 16; ++i) { float t = sv[i]; s += t; q += t * t; }
  float mean = s * (1.f / 16.f);
  float var  = q * (1.f / 16.f) - mean * mean;
  if (var < 0.f) var = 0.f;
  float o = (v - mean) * rsqrtf(var + BN_EPS) * gamma[co] + beta[co];
  out[(size_t)b * 512 + co] = o > 0.f ? o : 0.f;
}

// =======================  host orchestration  =======================
static inline int pad32i(int x) { return (x + 31) & ~31; }

extern "C" void kernel_launch(void* const* d_in, const int* in_sizes, int n_in,
                              void* d_out, int out_size, void* d_ws, size_t ws_size,
                              hipStream_t stream)
{
  (void)out_size; (void)ws_size;
  const int B = 16, N0 = 4096;

  // Locate xyz (unique flat size 16*4096*3 = 196608).
  int xi = -1;
  for (int i = 0; i < n_in; ++i) if (in_sizes[i] == B * N0 * 3) { xi = i; break; }
  if (xi < 0) xi = 0;

  // Params in flatten order (excluding xyz).
  int pidx[160]; int np = 0;
  for (int i = 0; i < n_in && np < 160; ++i) if (i != xi) pidx[np++] = i;
  // proj-first (jax sorted-key flatten) vs sa-first (insertion-order flatten)
  const bool proj_first = (np > 0 && in_sizes[pidx[0]] == 512 * 1536);
  int projBase, saBase[3];
  if (proj_first) { projBase = 0; saBase[0] = 4;  saBase[1] = 40; saBase[2] = 76; }
  else            { saBase[0] = 0; saBase[1] = 36; saBase[2] = 72; projBase = 108; }
  auto P = [&](int k) -> const float* { return (const float*)d_in[pidx[k]]; };

  static const int   mlps[3][3][3] = {
    {{32,32,64},{64,64,128},{64,96,128}},
    {{64,64,128},{128,128,256},{128,128,256}},
    {{256,256,512},{256,512,512},{256,512,512}}};
  static const float radii[3][3] = {{0.05f,0.1f,0.2f},{0.1f,0.2f,0.4f},{0.4f,0.8f,1.2f}};
  static const int   ks[3][3]    = {{16,32,64},{16,32,64},{32,64,128}};
  static const int   npoints[3]  = {512,128,1};
  static const int   Ns[3]       = {4096,512,128};
  static const int   Ctots[3]    = {320,640,1536};

  // Workspace bump allocator (256B aligned).
  size_t off = 0;
  auto alloc = [&](size_t bytes) -> void* {
    void* p = (void*)((char*)d_ws + off);
    off += (bytes + 255) & ~(size_t)255;
    return p;
  };
  float*  nx1 = (float*)alloc((size_t)B*512*3*4);
  float*  nx2 = (float*)alloc((size_t)B*128*3*4);
  float*  nx3 = (float*)alloc((size_t)B*1*3*4);
  float*  p1  = (float*)alloc((size_t)B*512*320*4);
  float*  p2  = (float*)alloc((size_t)B*128*640*4);
  float*  p3  = (float*)alloc((size_t)B*1*1536*4);
  int*    ballIdx = (int*)alloc((size_t)B*512*64*4);           // max B*S*K
  __bf16* Wb  = (__bf16*)alloc((size_t)512*672*2);             // max cout x cinPad
  float*  meanBuf = (float*)alloc(512*4);
  float*  rstdBuf = (float*)alloc(512*4);
  float*  yproj   = (float*)alloc((size_t)B*512*4);
  __bf16* Xbuf = (__bf16*)alloc((size_t)524288*96*2);          // max M x cinPad bf16
  float*  Ybuf = (float*)alloc((size_t)524288*128*4);          // max M x cout f32

  const float* cur_xyz = (const float*)d_in[xi];
  const float* cur_pts = nullptr;
  float* nxs[3] = {nx1, nx2, nx3};
  float* ps[3]  = {p1, p2, p3};
  int Cprev = 0;

  for (int lv = 0; lv < 3; ++lv) {
    const int N = Ns[lv], S = npoints[lv], Ctot = Ctots[lv];
    float* nxyz = nxs[lv];
    float* pout = ps[lv];

    fps_kernel<<<B, 1024, 0, stream>>>(cur_xyz, nxyz, N, S);

    int coff = 0;
    for (int br = 0; br < 3; ++br) {
      const int K = ks[lv][br];
      const float r = radii[lv][br];
      const int M = B * S * K;                    // always a multiple of 256
      const int cin0 = Cprev + 3;
      const int cinPad0 = pad32i(cin0);

      const int nWaves = B * S;
      const int bqBlocks = (nWaves * 32 + 255) / 256;
      ball_query_kernel<<<bqBlocks, 256, 0, stream>>>(
          cur_xyz, nxyz, ballIdx, B, N, S, K, r * r);
      build_group_kernel<<<M, 64, 0, stream>>>(
          cur_xyz, cur_pts, nxyz, ballIdx, Xbuf, N, S, K, Cprev, cinPad0);

      int cinL = cin0;
      for (int t = 0; t < 3; ++t) {
        const int cout = mlps[lv][br][t];
        const int Kp = pad32i(cinL);
        const int base = saBase[lv] + (br * 3 + t) * 4;
        const float* W     = P(base + 0);
        const float* bias  = P(base + 1);
        const float* gamma = P(base + 2);
        const float* beta  = P(base + 3);

        convert_w_kernel<<<cout, 128, 0, stream>>>(W, Wb, cinL, Kp);

        dim3 gg(M / 128, cout / 32);
        gemm_bf16_wmma<<<gg, 256, 0, stream>>>(Xbuf, Wb, bias, Ybuf, M, Kp, cout);

        stats_kernel<<<cout, 256, 0, stream>>>(Ybuf, M, cout, meanBuf, rstdBuf);

        if (t < 2) {
          size_t tot = (size_t)M * cout;          // next cin == cout (mult of 32)
          int gb = (int)((tot + 255) / 256);
          bn_relu_bf16_kernel<<<gb, 256, 0, stream>>>(
              Ybuf, meanBuf, rstdBuf, gamma, beta, Xbuf, tot, cout);
        } else {
          size_t tot = (size_t)B * S * cout;
          int gb = (int)((tot + 255) / 256);
          bn_relu_maxpool_kernel<<<gb, 256, 0, stream>>>(
              Ybuf, meanBuf, rstdBuf, gamma, beta, pout, B * S, K, cout, coff, Ctot);
        }
        cinL = cout;
      }
      coff += mlps[lv][br][2];
    }
    cur_xyz = nxyz;
    cur_pts = pout;
    Cprev = Ctot;
  }

  // Projection head: feats = p3 [16,1536] -> [16,512], BN over batch, ReLU.
  {
    const float* W     = P(projBase + 0);
    const float* bias  = P(projBase + 1);
    const float* gamma = P(projBase + 2);
    const float* beta  = P(projBase + 3);
    const int tot = B * 512;
    proj_gemm_kernel<<<(tot + 255) / 256, 256, 0, stream>>>(
        p3, W, bias, yproj, 1536, 512, tot);
    proj_bn_kernel<<<512, 16, 0, stream>>>(yproj, gamma, beta, (float*)d_out);
  }
}
